// SSD_cell_89051851916003
// MI455X (gfx1250) — matrix operand order repaired and checked by reading, not compiled, not run on hardware
//
#include <hip/hip_runtime.h>

// ---------------------------------------------------------------------------
// Mamba-2 / SSD chunked forward for MI455X (gfx1250, wave32, WMMA).
// All matmul-shaped work runs on v_wmma_f32_16x16x32_f16.
// ---------------------------------------------------------------------------

typedef __attribute__((ext_vector_type(16))) _Float16 v16h;
typedef __attribute__((ext_vector_type(8)))  float    v8f;

#define T_DIM   4096
#define B_DIM   4
#define S_DIM   16
#define F_DIM   64
#define N_DIM   16
#define P_DIM   64
#define K_TAP   5
#define CH_DIM  97
#define CHUNK   64
#define NCHUNK  64     // T / CHUNK
#define BS_DIM  64     // B * S
#define KTOT    320    // F * K_TAP, = 10 WMMA k-steps of 32
#define NT_CONV 7      // ceil(97/16)

__device__ __forceinline__ v8f wmma16(v16h a, v16h b, v8f c) {
  // D = A(16x32 f16) * B(32x16 f16) + C(16x16 f32)
  return __builtin_amdgcn_wmma_f32_16x16x32_f16(
      /*neg_a=*/false, a, /*neg_b=*/false, b,
      /*c_mod=*/(short)0, c, /*reuse_a=*/false, /*reuse_b=*/false);
}

// ---------------------------------------------------------------------------
// Kernel 0: pre-pack conv_w (CH,F,K) into WMMA B-operand fragment layout:
// [kstep][ntile][lane][half] f16, so the conv GEMM reads one v16h per lane.
// B layout: K = half + 16*(lane>=16) + 32*kstep ; N(ch) = lane%16 + 16*ntile.
// ---------------------------------------------------------------------------
__global__ void pack_w_kernel(const float* __restrict__ conv_w,
                              _Float16* __restrict__ wpack) {
  int idx = blockIdx.x * 256 + threadIdx.x;          // 10*7*32*16 = 35840
  if (idx >= 10 * NT_CONV * 512) return;
  int h    = idx & 15;
  int lane = (idx >> 4) & 31;
  int rest = idx >> 9;
  int nt   = rest % NT_CONV;
  int ks   = rest / NT_CONV;
  int ch = (lane & 15) + 16 * nt;
  int k  = h + ((lane >= 16) ? 16 : 0) + 32 * ks;
  float v = 0.f;
  if (ch < CH_DIM && k < KTOT) {
    int f = k / K_TAP, kk = k - K_TAP * f;
    v = conv_w[((size_t)ch * F_DIM + f) * K_TAP + kk];
  }
  wpack[idx] = (_Float16)v;
}

// ---------------------------------------------------------------------------
// Kernel 1: depth-conv (as GEMM, M=64 t-rows, N=112 ch, K=320) + bias +
// (sigmoid(z)+z), split into Bd/C/X (f16), lAd (f32), plus Dx->out.
// One workgroup per (bs, t-chunk); 4 waves, wave w owns t-rows [16w,16w+16).
// ---------------------------------------------------------------------------
__global__ __launch_bounds__(128)
void conv_wmma_kernel(const float* __restrict__ x,
                      const float* __restrict__ conv_b,
                      const float* __restrict__ pass_w,
                      const float* __restrict__ pass_b,
                      const float* __restrict__ red_b,
                      const float* __restrict__ dt_param,
                      const _Float16* __restrict__ wpack,
                      _Float16* __restrict__ Xh,
                      _Float16* __restrict__ Bdh,
                      _Float16* __restrict__ Ch,
                      float* __restrict__ lA,
                      float* __restrict__ out) {
  __shared__ __align__(16) _Float16 xs[68][64];   // x[t0-4 .. t0+63][f], f16
  __shared__ __align__(16) float    abuf[64][98]; // activated conv out
  __shared__ float coefb[64];
  __shared__ float dtsh;

  const int c  = blockIdx.x;
  const int bs = blockIdx.y;
  const int b  = bs >> 4, s = bs & 15;
  const int t0 = c * CHUNK;
  const int tid = threadIdx.x;

  if (tid == 0) dtsh = log1pf(expf(dt_param[0])) + 0.01f;   // softplus + bias

  for (int it = 0; it < 34; ++it) {                // 68*64 / 128
    int e = tid + it * 128;
    int j = e >> 6, f = e & 63;
    int tg = t0 - 4 + j;
    float v = (tg >= 0)
        ? x[(((size_t)tg * B_DIM + b) * S_DIM + s) * F_DIM + f] : 0.f;
    xs[j][f] = (_Float16)v;
  }
  __syncthreads();

  const int wave = tid >> 5, lane = tid & 31;
  const int mrow = (lane & 15) + 16 * wave;        // t-local row of A operand

  v8f acc[NT_CONV] = {};
  const v16h* wp = (const v16h*)wpack;
  for (int ks = 0; ks < 10; ++ks) {
    v16h afrag;
    #pragma unroll
    for (int h = 0; h < 16; ++h) {
      // A layout: K = h + 16*(lane>=16) + 32*ks
      int kidx = h + ((lane >= 16) ? 16 : 0) + 32 * ks;
      int f  = kidx / K_TAP;
      int kk = kidx - K_TAP * f;
      afrag[h] = xs[mrow + kk][f];                 // x[t0 + tl + kk - 4][f]
    }
    #pragma unroll
    for (int nt = 0; nt < NT_CONV; ++nt) {
      v16h bfrag = wp[(ks * NT_CONV + nt) * 32 + lane];
      acc[nt] = wmma16(afrag, bfrag, acc[nt]);
    }
  }

  // bias + (sigmoid(z)+z) -> abuf[t-local][ch]
  #pragma unroll
  for (int nt = 0; nt < NT_CONV; ++nt) {
    int ch = (lane & 15) + 16 * nt;
    if (ch < CH_DIM) {
      float cb = conv_b[ch];
      #pragma unroll
      for (int r = 0; r < 8; ++r) {
        int tl = 16 * wave + r + ((lane >= 16) ? 8 : 0);
        float z = acc[nt][r] + cb;
        abuf[tl][ch] = z + 1.f / (1.f + expf(-z));
      }
    }
  }
  __syncthreads();

  const float dt = dtsh;
  const size_t tbase = (((size_t)b * NCHUNK + c) * S_DIM + s);
  if (tid < 64) {
    int tl = tid;
    float ar  = abuf[tl][2 * N_DIM];
    float A   = -fabsf(ar);
    float lad = dt * A;
    coefb[tl] = expm1f(lad) / (A + 1e-9f);         // (exp(dt*A)-1)/(A+eps)
    lA[((size_t)(b * S_DIM + s) * NCHUNK + c) * CHUNK + tl] = lad;
    // Dx pass-through + leaky relu + red_b  (written fresh every launch)
    float dx = pass_b[0];
    #pragma unroll
    for (int f = 0; f < F_DIM; ++f) dx += (float)xs[tl + 4][f] * pass_w[f];
    float dxa = (dx >= 0.f) ? dx : 0.01f * dx;
    out[(((size_t)(t0 + tl)) * B_DIM + b) * S_DIM + s] = red_b[0] + dxa;
  }
  __syncthreads();

  #pragma unroll
  for (int it = 0; it < 8; ++it) {                 // Bd, C : 64x16
    int e = tid + it * 128;
    int tl = e >> 4, n = e & 15;
    size_t gi = tbase * 1024 + (size_t)tl * 16 + n;
    Bdh[gi] = (_Float16)(abuf[tl][n] * coefb[tl]);
    Ch[gi]  = (_Float16)(abuf[tl][N_DIM + n]);
  }
  #pragma unroll
  for (int it = 0; it < 32; ++it) {                // X : 64x64
    int e = tid + it * 128;
    int tl = e >> 6, p = e & 63;
    Xh[tbase * 4096 + (size_t)tl * 64 + p] = (_Float16)(abuf[tl][2 * N_DIM + 1 + p]);
  }
}

// ---------------------------------------------------------------------------
// Kernel 2: per-(b,s,c) cumsum of lAd along chunk + chunk totals.
// ---------------------------------------------------------------------------
__global__ void cumsum_kernel(const float* __restrict__ lA,
                              float* __restrict__ Acum,
                              float* __restrict__ Alast) {
  int idx = blockIdx.x * 256 + threadIdx.x;        // B*S*NCHUNK = 4096
  if (idx >= BS_DIM * NCHUNK) return;
  const float* src = lA   + (size_t)idx * CHUNK;
  float*       dst = Acum + (size_t)idx * CHUNK;
  float ssum = 0.f;
  for (int l = 0; l < CHUNK; ++l) { ssum += src[l]; dst[l] = ssum; }
  Alast[idx] = ssum;
}

// ---------------------------------------------------------------------------
// Kernel 3: chunk states = X^T (64x64) @ (Bd * decay_states) (64x16) via WMMA.
// ---------------------------------------------------------------------------
__global__ __launch_bounds__(128)
void chunk_states_kernel(const _Float16* __restrict__ Xh,
                         const _Float16* __restrict__ Bdh,
                         const float* __restrict__ Acum,
                         const float* __restrict__ Alast,
                         float* __restrict__ states) {
  __shared__ __align__(16) _Float16 xsh[64][64];
  __shared__ __align__(16) _Float16 bdec[64][16];
  __shared__ float ac[64];

  const int c = blockIdx.x, bs = blockIdx.y;
  const int b = bs >> 4, s = bs & 15;
  const int tid = threadIdx.x;
  const size_t tbase = (((size_t)b * NCHUNK + c) * S_DIM + s);

  const unsigned long long* xg = (const unsigned long long*)(Xh + tbase * 4096);
  unsigned long long* xd = (unsigned long long*)xsh;
  #pragma unroll
  for (int it = 0; it < 8; ++it) xd[tid + it * 128] = xg[tid + it * 128];
  if (tid < 64)
    ac[tid] = Acum[((size_t)(b * S_DIM + s) * NCHUNK + c) * CHUNK + tid];
  __syncthreads();

  const float al = Alast[(size_t)(b * S_DIM + s) * NCHUNK + c];
  const _Float16* bg = Bdh + tbase * 1024;
  #pragma unroll
  for (int it = 0; it < 8; ++it) {
    int e = tid + it * 128;
    int l = e >> 4, n = e & 15;
    float d = expf(fmaxf(-20.f, al - ac[l]));      // decay_states[l]
    bdec[l][n] = (_Float16)((float)bg[e] * d);
  }
  __syncthreads();

  const int wave = tid >> 5, lane = tid & 31;
  v8f acc = {};
  #pragma unroll
  for (int ks = 0; ks < 2; ++ks) {
    v16h a, bb;
    #pragma unroll
    for (int h = 0; h < 16; ++h) {                 // A = X^T : M=p, K=l
      int l = h + ((lane >= 16) ? 16 : 0) + 32 * ks;
      a[h] = xsh[l][(lane & 15) + 16 * wave];
    }
    #pragma unroll
    for (int h = 0; h < 16; ++h) {                 // B = Bdec : K=l, N=n
      int l = h + ((lane >= 16) ? 16 : 0) + 32 * ks;
      bb[h] = bdec[l][lane & 15];
    }
    acc = wmma16(a, bb, acc);
  }
  #pragma unroll
  for (int r = 0; r < 8; ++r) {
    int p = 16 * wave + r + ((lane >= 16) ? 8 : 0);
    states[tbase * 1024 + (size_t)p * 16 + (lane & 15)] = acc[r];
  }
}

// ---------------------------------------------------------------------------
// Kernel 4: inter-chunk decay scan (exact recurrence form of decay_chunk@states)
// carried[z] = states_c[z];  carried <- exp(clip(Alast[z]))*carried + states[z]
// ---------------------------------------------------------------------------
__global__ __launch_bounds__(256)
void scan_kernel(const float* __restrict__ states,
                 const float* __restrict__ Alast,
                 _Float16* __restrict__ Stc) {
  const int bs = blockIdx.x;
  const int b = bs >> 4, s = bs & 15;
  const int tid = threadIdx.x;
  float carried[4] = {0.f, 0.f, 0.f, 0.f};
  const float* al = Alast + (size_t)bs * NCHUNK;
  for (int z = 0; z < NCHUNK; ++z) {
    size_t base = (((size_t)b * NCHUNK + z) * S_DIM + s) * 1024;
    float dec = expf(fmaxf(-20.f, al[z]));
    #pragma unroll
    for (int k = 0; k < 4; ++k) {
      int e = tid + k * 256;                       // e = p*16 + n
      Stc[base + e] = (_Float16)carried[k];        // state entering chunk z
      carried[k] = dec * carried[k] + states[base + e];
    }
  }
}

// ---------------------------------------------------------------------------
// Kernel 5: per-(b,c,s) tile:
//   G = Cb @ Bd^T            (WMMA, K=16 padded to 32)
//   GL = G .* exp(Acum[l]-Acum[m]) lower-tri
//   Y  = (sdo*Cb) @ states_c^T  +  GL @ X     (WMMA)
//   out += Y @ red_w
// ---------------------------------------------------------------------------
__global__ __launch_bounds__(128)
void ssd_tile_kernel(const _Float16* __restrict__ Xh,
                     const _Float16* __restrict__ Bdh,
                     const _Float16* __restrict__ Ch,
                     const _Float16* __restrict__ Stc,
                     const float* __restrict__ Acum,
                     const float* __restrict__ red_w,
                     float* __restrict__ out) {
  __shared__ __align__(16) _Float16 xsh[64][64];
  __shared__ __align__(16) _Float16 bdh[64][16];
  __shared__ __align__(16) _Float16 cbh[64][16];
  __shared__ __align__(16) _Float16 stc[64][16];
  __shared__ __align__(16) _Float16 gl[64][64];
  __shared__ float ac[64];
  __shared__ float rsw[64];
  __shared__ float ylds[64];

  const int c = blockIdx.x, bs = blockIdx.y;
  const int b = bs >> 4, s = bs & 15;
  const int tid = threadIdx.x;
  const size_t tbase = (((size_t)b * NCHUNK + c) * S_DIM + s);

  {
    const unsigned long long* xg = (const unsigned long long*)(Xh + tbase * 4096);
    unsigned long long* xd = (unsigned long long*)xsh;
    #pragma unroll
    for (int it = 0; it < 8; ++it) xd[tid + it * 128] = xg[tid + it * 128];
    ((unsigned long long*)bdh)[tid] = ((const unsigned long long*)(Bdh + tbase * 1024))[tid];
    ((unsigned long long*)cbh)[tid] = ((const unsigned long long*)(Ch  + tbase * 1024))[tid];
    ((unsigned long long*)stc)[tid] = ((const unsigned long long*)(Stc + tbase * 1024))[tid];
    if (tid < 64) {
      ac[tid]  = Acum[((size_t)(b * S_DIM + s) * NCHUNK + c) * CHUNK + tid];
      rsw[tid] = red_w[tid];
      ylds[tid] = 0.f;
    }
  }
  __syncthreads();

  const int wave = tid >> 5, lane = tid & 31;
  const int lrow = 16 * wave + (lane & 15);        // A-operand row (t-local)

  // A = Cb : M=l, K=n (n in [0,16), padded to 32 with zeros)
  v16h a_cb;
  #pragma unroll
  for (int h = 0; h < 16; ++h) {
    int n = h + ((lane >= 16) ? 16 : 0);
    a_cb[h] = (n < 16) ? cbh[lrow][n] : (_Float16)0.f;
  }
  // G = Cb @ Bd^T   (B: K=n, N=m; rows K>=16 are zero -> lanes>=16 zero)
  v8f g[4];
  #pragma unroll
  for (int nt = 0; nt < 4; ++nt) {
    v16h bb;
    #pragma unroll
    for (int h = 0; h < 16; ++h) {
      int m = (lane & 15) + 16 * nt;
      bb[h] = (lane < 16) ? bdh[m][h] : (_Float16)0.f;
    }
    v8f z = {};
    g[nt] = wmma16(a_cb, bb, z);
  }
  // mask with causal decay L and restage as f16 A-operand source in LDS
  #pragma unroll
  for (int nt = 0; nt < 4; ++nt) {
    #pragma unroll
    for (int r = 0; r < 8; ++r) {
      int l = 16 * wave + r + ((lane >= 16) ? 8 : 0);
      int m = (lane & 15) + 16 * nt;
      float v = g[nt][r];
      v = (l >= m) ? v * expf(fmaxf(-20.f, ac[l] - ac[m])) : 0.f;
      gl[l][m] = (_Float16)v;
    }
  }

  // Y_off: A = sdo[l]*Cb, B = states_c^T (K=n, N=p)
  const float sdo = expf(ac[lrow]);                // state_decay_out[l]
  v16h a_cs;
  #pragma unroll
  for (int h = 0; h < 16; ++h) {
    int n = h + ((lane >= 16) ? 16 : 0);
    a_cs[h] = (n < 16) ? (_Float16)((float)cbh[lrow][n] * sdo) : (_Float16)0.f;
  }
  v8f y[4];
  #pragma unroll
  for (int nt = 0; nt < 4; ++nt) {
    v16h bb;
    #pragma unroll
    for (int h = 0; h < 16; ++h) {
      int p = (lane & 15) + 16 * nt;
      bb[h] = (lane < 16) ? stc[p][h] : (_Float16)0.f;
    }
    v8f z = {};
    y[nt] = wmma16(a_cs, bb, z);
  }
  // Y_diag: (G.*L) @ X   (K=m over 64, 2 k-steps)
  #pragma unroll
  for (int ks = 0; ks < 2; ++ks) {
    v16h ag;
    #pragma unroll
    for (int h = 0; h < 16; ++h) {
      int m = h + ((lane >= 16) ? 16 : 0) + 32 * ks;
      ag[h] = gl[lrow][m];
    }
    #pragma unroll
    for (int nt = 0; nt < 4; ++nt) {
      v16h bx;
      #pragma unroll
      for (int h = 0; h < 16; ++h) {
        int m = h + ((lane >= 16) ? 16 : 0) + 32 * ks;
        bx[h] = xsh[m][(lane & 15) + 16 * nt];
      }
      y[nt] = wmma16(ag, bx, y[nt]);
    }
  }

  // y[l] = sum_p Y[l][p] * red_w[p]  (lane-reduce via LDS f32 atomics)
  #pragma unroll
  for (int r = 0; r < 8; ++r) {
    int l = 16 * wave + r + ((lane >= 16) ? 8 : 0);
    float part = 0.f;
    #pragma unroll
    for (int nt = 0; nt < 4; ++nt)
      part += y[nt][r] * rsw[(lane & 15) + 16 * nt];
    atomicAdd(&ylds[l], part);
  }
  __syncthreads();
  if (tid < 64) {
    size_t t = (size_t)c * CHUNK + tid;
    size_t oi = (t * B_DIM + b) * S_DIM + s;
    out[oi] += ylds[tid];                          // out holds red_b + leaky(Dx)
  }
}

// ---------------------------------------------------------------------------
extern "C" void kernel_launch(void* const* d_in, const int* in_sizes, int n_in,
                              void* d_out, int out_size, void* d_ws, size_t ws_size,
                              hipStream_t stream) {
  (void)in_sizes; (void)n_in; (void)out_size; (void)ws_size;
  const float* x        = (const float*)d_in[0];
  const float* conv_w   = (const float*)d_in[1];
  const float* conv_b   = (const float*)d_in[2];
  const float* pass_w   = (const float*)d_in[3];
  const float* pass_b   = (const float*)d_in[4];
  const float* red_w    = (const float*)d_in[5];
  const float* red_b    = (const float*)d_in[6];
  const float* dt_param = (const float*)d_in[7];
  float* out = (float*)d_out;

  char* base = (char*)d_ws;
  size_t off = 0;
  auto take = [&](size_t bytes) {
    char* p = base + off;
    off += (bytes + 255) & ~(size_t)255;
    return (void*)p;
  };
  _Float16* wpack  = (_Float16*)take((size_t)10 * NT_CONV * 512 * 2);          // 70 KB
  _Float16* Xh     = (_Float16*)take((size_t)BS_DIM * NCHUNK * 4096 * 2);      // 32 MB
  _Float16* Bdh    = (_Float16*)take((size_t)BS_DIM * NCHUNK * 1024 * 2);      // 8 MB
  _Float16* Chm    = (_Float16*)take((size_t)BS_DIM * NCHUNK * 1024 * 2);      // 8 MB
  _Float16* Stc    = (_Float16*)take((size_t)BS_DIM * NCHUNK * 1024 * 2);      // 8 MB
  float*    lAbuf  = (float*)   take((size_t)BS_DIM * NCHUNK * CHUNK * 4);     // 1 MB
  float*    Acum   = (float*)   take((size_t)BS_DIM * NCHUNK * CHUNK * 4);     // 1 MB
  float*    Alast  = (float*)   take((size_t)BS_DIM * NCHUNK * 4);             // 16 KB
  float*    states = (float*)   take((size_t)BS_DIM * NCHUNK * 1024 * 4);      // 16 MB

  pack_w_kernel<<<(10 * NT_CONV * 512 + 255) / 256, 256, 0, stream>>>(conv_w, wpack);

  conv_wmma_kernel<<<dim3(NCHUNK, BS_DIM), 128, 0, stream>>>(
      x, conv_b, pass_w, pass_b, red_b, dt_param, wpack,
      Xh, Bdh, Chm, lAbuf, out);

  cumsum_kernel<<<(BS_DIM * NCHUNK + 255) / 256, 256, 0, stream>>>(lAbuf, Acum, Alast);

  chunk_states_kernel<<<dim3(NCHUNK, BS_DIM), 128, 0, stream>>>(
      Xh, Bdh, Acum, Alast, states);

  scan_kernel<<<BS_DIM, 256, 0, stream>>>(states, Alast, Stc);

  ssd_tile_kernel<<<dim3(NCHUNK, BS_DIM), 128, 0, stream>>>(
      Xh, Bdh, Chm, Stc, Acum, red_w, out);
}
